// Sparse_Spatial_AttentionCOPY_7584912244782
// MI455X (gfx1250) — matrix operand order repaired and checked by reading, not compiled, not run on hardware
//
#include <hip/hip_runtime.h>

typedef __attribute__((ext_vector_type(16))) __bf16 v16bf;
typedef __attribute__((ext_vector_type(8)))  float  v8f;
typedef __attribute__((ext_vector_type(4)))  unsigned v4u;
typedef __attribute__((ext_vector_type(8)))  int   v8i;
typedef __attribute__((ext_vector_type(4)))  int   v4i;

#define DEV static __device__ __forceinline__

constexpr int Bb = 4, Tt = 12, Nn = 1024, Kk = 16, Hh = 4, Dd = 32, Ff = 128;
constexpr int Ss = 50, Sp = 64;            // top-k count, padded to WMMA tiles
constexpr int R   = Bb * Tt * Nn;          // 49152 token rows
constexpr int HBT = Hh * Bb * Tt;          // 192 attention "rows"
constexpr float SCALE = 0.17677669529663687f; // 1/sqrt(32)

// --- WMMA fragment index helpers (CDNA5 ISA 7.12.2, wave32) ---
// A 16x32 bf16: lane L<16 -> row M=L, elems j: K={0..7,16..23}; L>=16: K={8..15,24..31}
DEV int a_k(int lane, int j) { return ((j < 8) ? j : j + 8) + ((lane >> 4) << 3); }
// B 32x16 bf16: col N = lane&15; lanes 0-15 hold K={0..7,16..23}, lanes 16-31 hold K={8..15,24..31}
DEV int b_k(int lane, int j) { return ((j < 8) ? j : j + 8) + ((lane >> 4) << 3); }

DEV v8f wmma_bf16(v16bf a, v16bf b, v8f c) {
  return __builtin_amdgcn_wmma_f32_16x16x32_bf16(false, a, false, b, (short)0, c,
                                                 false, false);
}

// TDM: async DMA of a rows x 128-float 2D tile (row stride 128) into LDS.
// D# packing per CDNA5 ISA ch.8: group0 = count/lds/global/type, group1 = dims.
// This toolchain exposes the 6-arg builtin (groups 2/3 + extra group zeroed).
DEV void tdm_load_tile_f32(const float* __restrict__ src, float* ldsdst,
                           int rows) {
  unsigned lds = (unsigned)(unsigned long long)(void*)ldsdst; // flat->LDS offset
  unsigned long long ga = (unsigned long long)(const void*)src;
  v4u g0;
  g0.x = 1u;                                   // count=1, user descriptor
  g0.y = lds;                                  // lds_addr
  g0.z = (unsigned)ga;                         // global_addr[31:0]
  g0.w = (unsigned)((ga >> 32) & 0x01ffffffu) | 0x80000000u; // addr[56:32]|type=2
  v8i g1;
  g1[0] = 0x20000;                             // data_size=2 (4 bytes)
  g1[1] = (int)((unsigned)(Ff & 0xffff) << 16);            // tensor_dim0 lo16
  g1[2] = (int)((unsigned)(Ff >> 16) | ((unsigned)(R & 0xffff) << 16)); // td0 hi|td1 lo
  g1[3] = (int)((unsigned)(R >> 16) | ((unsigned)Ff << 16));  // td1 hi | tile_dim0=128
  g1[4] = rows;                                // tile_dim1 | tile_dim2=0
  g1[5] = Ff;                                  // tensor_dim0_stride lo32
  g1[6] = 0;                                   // stride0 hi | stride1 lo
  g1[7] = 0;                                   // stride1 hi
  v4i z4 = {};
  v8i z8 = {};
  __builtin_amdgcn_tensor_load_to_lds(g0, g1, z4, z4, z8, 0);
}

// ---------------------------------------------------------------------------
// K1: x_ = x + spa_vec*spa_val + tem_vec*tem_val, then Q/K/V = x_ @ W^T + b
// ---------------------------------------------------------------------------
__global__ __launch_bounds__(256) void k_embed_qkv(
    const float* __restrict__ x, const float* __restrict__ sval,
    const float* __restrict__ svec, const float* __restrict__ tval,
    const float* __restrict__ tvec, const float* __restrict__ Wq,
    const float* __restrict__ bq, const float* __restrict__ Wk,
    const float* __restrict__ bk, const float* __restrict__ Wv,
    const float* __restrict__ bv, float* __restrict__ Qp,
    float* __restrict__ Kp, float* __restrict__ Vp) {
  __shared__ __bf16 xs[16 * Ff];
  const int rt = blockIdx.x, tid = threadIdx.x;
#pragma unroll
  for (int i = 0; i < 8; ++i) {
    int e = tid + i * 256;                 // 16*128 = 2048 elems
    int rl = e >> 7, f = e & 127;
    int r = rt * 16 + rl;
    int n = r & (Nn - 1);
    float v = x[(size_t)r * Ff + f] + svec[(size_t)n * Ff + f] * sval[f] +
              tvec[(size_t)n * Ff + f] * tval[f];
    xs[rl * Ff + f] = (__bf16)v;
  }
  __syncthreads();
  const int wave = tid >> 5, lane = tid & 31;
  const int col = lane & 15, hf = lane >> 4;
  const int ct = wave;
  v16bf afrag[4];
#pragma unroll
  for (int kc = 0; kc < 4; ++kc)
#pragma unroll
    for (int j = 0; j < 16; ++j)
      afrag[kc][j] = xs[col * Ff + kc * 32 + a_k(lane, j)];
  const float* Ws[3] = {Wq, Wk, Wv};
  const float* bs[3] = {bq, bk, bv};
  float* Os[3] = {Qp, Kp, Vp};
#pragma unroll
  for (int m = 0; m < 3; ++m) {
    const float* W = Ws[m];
    v8f acc = {};
#pragma unroll
    for (int kc = 0; kc < 4; ++kc) {
      v16bf bfr;
#pragma unroll
      for (int j = 0; j < 16; ++j)
        bfr[j] = (__bf16)W[(size_t)(ct * 16 + col) * Ff + kc * 32 + b_k(lane, j)];
      acc = wmma_bf16(afrag[kc], bfr, acc);
    }
    float* O = Os[m];
    int c = ct * 16 + col;
#pragma unroll
    for (int g = 0; g < 8; ++g) {
      int rr = rt * 16 + g + hf * 8;
      O[(size_t)rr * Ff + c] = acc[g] + bs[m][c];
    }
  }
}

// ---------------------------------------------------------------------------
// K2: qk_sample + learned score M.  One wave per (h,b,t,n); lane = d.
// ---------------------------------------------------------------------------
__global__ __launch_bounds__(256) void k_qk_m(
    const float* __restrict__ Qp, const float* __restrict__ Kp,
    const int* __restrict__ adj, const float* __restrict__ Wp,
    const float* __restrict__ bp, float* __restrict__ M) {
  int wid = (blockIdx.x * 256 + threadIdx.x) >> 5;
  int lane = threadIdx.x & 31;
  int n = wid & (Nn - 1);
  int t = (wid >> 10) % Tt;
  int b = (wid / (Nn * Tt)) & 3;
  int h = wid / (Nn * Tt * Bb);
  size_t bt = (size_t)(b * Tt + t) * Nn;
  float q = Qp[(bt + n) * Ff + h * Dd + lane];
  float m = 0.f;
#pragma unroll
  for (int k = 0; k < Kk; ++k) {
    int j = adj[n * Kk + k];
    float p = q * Kp[(bt + j) * Ff + h * Dd + lane];
#pragma unroll
    for (int off = 16; off; off >>= 1) p += __shfl_xor(p, off, 32);
    m += p * Wp[k];
  }
  if (lane == 0) {
    int hbt = (h * Bb + b) * Tt + t;
    M[(size_t)hbt * Nn + n] = m + bp[0];
  }
}

// ---------------------------------------------------------------------------
// K3: iterative top-50 (argmax + knockout), matches lax.top_k ordering.
// ---------------------------------------------------------------------------
__global__ __launch_bounds__(256) void k_topk(const float* __restrict__ M,
                                              int* __restrict__ Mtop) {
  __shared__ float sv[Nn];
  __shared__ float rv[256];
  __shared__ int ri[256];
  int hbt = blockIdx.x, tid = threadIdx.x;
  for (int i = tid; i < Nn; i += 256) sv[i] = M[(size_t)hbt * Nn + i];
  __syncthreads();
  for (int s = 0; s < Ss; ++s) {
    float bvv = -INFINITY;
    int bi = Nn;
    for (int i = tid; i < Nn; i += 256) {
      float v = sv[i];
      if (v > bvv) { bvv = v; bi = i; }
    }
    rv[tid] = bvv; ri[tid] = bi;
    __syncthreads();
    for (int off = 128; off; off >>= 1) {
      if (tid < off) {
        float ov = rv[tid + off]; int oi = ri[tid + off];
        if (ov > rv[tid] || (ov == rv[tid] && oi < ri[tid])) {
          rv[tid] = ov; ri[tid] = oi;
        }
      }
      __syncthreads();
    }
    if (tid == 0) { Mtop[hbt * Ss + s] = ri[0]; sv[ri[0]] = -INFINITY; }
    __syncthreads();
  }
}

// ---------------------------------------------------------------------------
// K4: scores = Q_reduce @ K^T * SCALE via WMMA (K-dim = D = 32, one WMMA/tile)
// ---------------------------------------------------------------------------
__global__ __launch_bounds__(256) void k_scores(
    const float* __restrict__ Qp, const float* __restrict__ Kp,
    const int* __restrict__ Mtop, float* __restrict__ Sc) {
  int hbt = blockIdx.x;
  int t = hbt % Tt, b = (hbt / Tt) % Bb, h = hbt / (Tt * Bb);
  int stile = blockIdx.y;
  int wave = threadIdx.x >> 5, lane = threadIdx.x & 31;
  int ntile = blockIdx.z * 8 + wave;
  int col = lane & 15, hf = lane >> 4;
  size_t bt = (size_t)(b * Tt + t) * Nn;
  v16bf a;
  int s = stile * 16 + col;
  if (s < Ss) {
    const float* qrow = Qp + (bt + Mtop[hbt * Ss + s]) * Ff + h * Dd;
#pragma unroll
    for (int j = 0; j < 16; ++j) a[j] = (__bf16)qrow[a_k(lane, j)];
  } else {
#pragma unroll
    for (int j = 0; j < 16; ++j) a[j] = (__bf16)0.f;
  }
  int n = ntile * 16 + col;
  const float* krow = Kp + (bt + n) * Ff + h * Dd;
  v16bf bfr;
#pragma unroll
  for (int j = 0; j < 16; ++j) bfr[j] = (__bf16)krow[b_k(lane, j)];
  v8f acc = {};
  acc = wmma_bf16(a, bfr, acc);
#pragma unroll
  for (int g = 0; g < 8; ++g) {
    int sr = stile * 16 + g + hf * 8;
    Sc[((size_t)hbt * Sp + sr) * Nn + n] = acc[g] * SCALE;
  }
}

// K5: softmax over n (one block per padded s-row; pad rows are all-zero -> safe)
__global__ __launch_bounds__(256) void k_softmax(float* __restrict__ Sc) {
  __shared__ float red[256];
  size_t base = (size_t)blockIdx.x * Nn;
  int tid = threadIdx.x;
  float mx = -INFINITY;
  for (int i = tid; i < Nn; i += 256) mx = fmaxf(mx, Sc[base + i]);
  red[tid] = mx; __syncthreads();
  for (int off = 128; off; off >>= 1) {
    if (tid < off) red[tid] = fmaxf(red[tid], red[tid + off]);
    __syncthreads();
  }
  mx = red[0]; __syncthreads();
  float sum = 0.f;
  for (int i = tid; i < Nn; i += 256) {
    float e = __expf(Sc[base + i] - mx);
    Sc[base + i] = e; sum += e;
  }
  red[tid] = sum; __syncthreads();
  for (int off = 128; off; off >>= 1) {
    if (tid < off) red[tid] += red[tid + off];
    __syncthreads();
  }
  float inv = 1.f / red[0];
  for (int i = tid; i < Nn; i += 256) Sc[base + i] *= inv;
}

// K6: cp = argmax over s (first 50 rows only, first-hit tie-break)
__global__ __launch_bounds__(256) void k_argmax(const float* __restrict__ Sc,
                                                int* __restrict__ Cp) {
  int idx = blockIdx.x * 256 + threadIdx.x;   // hbt*Nn + n
  int hbt = idx >> 10, n = idx & (Nn - 1);
  float bvv = -INFINITY; int bs = 0;
  for (int s = 0; s < Ss; ++s) {
    float v = Sc[((size_t)hbt * Sp + s) * Nn + n];
    if (v > bvv) { bvv = v; bs = s; }
  }
  Cp[idx] = bs;
}

// ---------------------------------------------------------------------------
// K7: av = attn @ V via WMMA, split-K: 4 blocks of 8 K-chunks each -> 4 partials
// (deterministic tree in k_gather; avoids float atomics). 8 waves = 4x2 tiles.
// ---------------------------------------------------------------------------
__global__ __launch_bounds__(256) void k_av(const float* __restrict__ Sc,
                                            const float* __restrict__ Vp,
                                            float* __restrict__ Av4) {
  int hbt = blockIdx.x;
  int kg = blockIdx.y;                      // split-K group: 0..3
  int t = hbt % Tt, b = (hbt / Tt) % Bb, h = hbt / (Tt * Bb);
  int wave = threadIdx.x >> 5, lane = threadIdx.x & 31;
  int stile = wave >> 1, dtile = wave & 1;
  int col = lane & 15, hf = lane >> 4;
  size_t bt = (size_t)(b * Tt + t) * Nn;
  const float* arow = Sc + ((size_t)hbt * Sp + stile * 16 + col) * Nn;
  v8f acc = {};
  for (int kc = kg * 8; kc < kg * 8 + 8; ++kc) {
    v16bf a, bfr;
#pragma unroll
    for (int j = 0; j < 16; ++j) a[j] = (__bf16)arow[kc * 32 + a_k(lane, j)];
    int d = dtile * 16 + col;
#pragma unroll
    for (int j = 0; j < 16; ++j) {
      int n = kc * 32 + b_k(lane, j);
      bfr[j] = (__bf16)Vp[(bt + n) * Ff + h * Dd + d];
    }
    acc = wmma_bf16(a, bfr, acc);
  }
#pragma unroll
  for (int g = 0; g < 8; ++g) {
    int sr = stile * 16 + g + hf * 8;
    Av4[(((size_t)kg * HBT + hbt) * Sp + sr) * Dd + dtile * 16 + col] = acc[g];
  }
}

// K8: value = gather av rows at cp (summing 4 split-K partials), merge heads
__global__ __launch_bounds__(256) void k_gather(const float* __restrict__ Av4,
                                                const int* __restrict__ Cp,
                                                float* __restrict__ Val) {
  int idx = blockIdx.x * 256 + threadIdx.x;   // r*Ff + f
  int f = idx & 127, r = idx >> 7;
  int n = r & (Nn - 1);
  int t = (r >> 10) % Tt;
  int b = r / (Nn * Tt);
  int h = f >> 5, d = f & 31;
  int hbt = (h * Bb + b) * Tt + t;
  int cp = Cp[hbt * Nn + n];
  float v = 0.f;
#pragma unroll
  for (int g = 0; g < 4; ++g)
    v += Av4[(((size_t)g * HBT + hbt) * Sp + cp) * Dd + d];
  Val[idx] = v;
}

// ---------------------------------------------------------------------------
// K9: generic 128x128 GEMM  O = X @ W^T + bias (+epilogue).
// X tile is DMA'd into LDS by the Tensor Data Mover (async, TENSORcnt), awaited before
// waves fetch weight fragments from global; bf16 convert at fragment build.
// ---------------------------------------------------------------------------
template <int EPI>
__global__ __launch_bounds__(256) void k_gemm(const float* __restrict__ X,
                                              const float* __restrict__ W,
                                              const float* __restrict__ bias,
                                              const float* __restrict__ res,
                                              float* __restrict__ O) {
  __shared__ float xsf[16 * Ff];
  int rt = blockIdx.x, tid = threadIdx.x;
  if (tid < 32) {                      // wave 0 issues the tensor DMA
    tdm_load_tile_f32(X + (size_t)rt * 16 * Ff, xsf, 16);
    __builtin_amdgcn_s_wait_tensorcnt(0);
  }
  __syncthreads();
  int wave = tid >> 5, lane = tid & 31;
  int col = lane & 15, hf = lane >> 4;
  int ct = wave;
  v8f acc = {};
#pragma unroll
  for (int kc = 0; kc < 4; ++kc) {
    v16bf a, bfr;
#pragma unroll
    for (int j = 0; j < 16; ++j)
      a[j] = (__bf16)xsf[col * Ff + kc * 32 + a_k(lane, j)];
#pragma unroll
    for (int j = 0; j < 16; ++j)
      bfr[j] = (__bf16)W[(size_t)(ct * 16 + col) * Ff + kc * 32 + b_k(lane, j)];
    acc = wmma_bf16(a, bfr, acc);
  }
  int c = ct * 16 + col;
#pragma unroll
  for (int g = 0; g < 8; ++g) {
    size_t rr = (size_t)rt * 16 + g + hf * 8;
    float v = acc[g] + bias[c];
    if constexpr (EPI == 1) v = fmaxf(v, 0.f);
    if constexpr (EPI == 2) v += res[rr * Ff + c];
    O[rr * Ff + c] = v;
  }
}

// K10: layernorm over F=128 (one wave per row, 4 elems/lane)
__global__ __launch_bounds__(256) void k_ln(const float* __restrict__ X,
                                            const float* __restrict__ g,
                                            const float* __restrict__ bta,
                                            float* __restrict__ O, int affine) {
  int wid = (blockIdx.x * 256 + threadIdx.x) >> 5;
  int lane = threadIdx.x & 31;
  const float* xr = X + (size_t)wid * Ff;
  float v[4], s = 0.f;
#pragma unroll
  for (int i = 0; i < 4; ++i) { v[i] = xr[lane * 4 + i]; s += v[i]; }
#pragma unroll
  for (int off = 16; off; off >>= 1) s += __shfl_xor(s, off, 32);
  float mu = s * (1.f / Ff), q = 0.f;
#pragma unroll
  for (int i = 0; i < 4; ++i) { float dv = v[i] - mu; q += dv * dv; }
#pragma unroll
  for (int off = 16; off; off >>= 1) q += __shfl_xor(q, off, 32);
  float rstd = rsqrtf(q * (1.f / Ff) + 1e-5f);
  float* orow = O + (size_t)wid * Ff;
#pragma unroll
  for (int i = 0; i < 4; ++i) {
    int c = lane * 4 + i;
    float y = (v[i] - mu) * rstd;
    orow[c] = affine ? (y * g[c] + bta[c]) : y;
  }
}

extern "C" void kernel_launch(void* const* d_in, const int* in_sizes, int n_in,
                              void* d_out, int out_size, void* d_ws,
                              size_t ws_size, hipStream_t stream) {
  const float* x    = (const float*)d_in[0];
  const float* sval = (const float*)d_in[1];
  const float* svec = (const float*)d_in[2];
  const float* tval = (const float*)d_in[3];
  const float* tvec = (const float*)d_in[4];
  const float* Wq = (const float*)d_in[5];  const float* bq = (const float*)d_in[6];
  const float* Wk = (const float*)d_in[7];  const float* bk = (const float*)d_in[8];
  const float* Wv = (const float*)d_in[9];  const float* bv = (const float*)d_in[10];
  const float* Wp = (const float*)d_in[11]; const float* bp = (const float*)d_in[12];
  const float* Wo = (const float*)d_in[13]; const float* bo = (const float*)d_in[14];
  const float* lng = (const float*)d_in[15]; const float* lnb = (const float*)d_in[16];
  const float* W1 = (const float*)d_in[17]; const float* b1 = (const float*)d_in[18];
  const float* W2 = (const float*)d_in[19]; const float* b2 = (const float*)d_in[20];
  const int* adj = (const int*)d_in[21];
  float* out = (float*)d_out;

  // workspace carve-up (floats); Q/K/V & attn regions recycled for the FFN
  float* Qp = (float*)d_ws;
  float* Kp = Qp + (size_t)R * Ff;
  float* Vp = Kp + (size_t)R * Ff;
  float* Mb = Vp + (size_t)R * Ff;
  int* Mtop = (int*)(Mb + (size_t)HBT * Nn);
  float* Sc = (float*)(Mtop + HBT * Ss);
  int* Cp = (int*)(Sc + (size_t)HBT * Sp * Nn);
  float* Av4 = (float*)(Cp + HBT * Nn);       // 4 split-K partials
  float* Val0 = Qp;   // dead after k_scores
  float* Tmp1 = Kp;   // dead after k_scores
  float* Val1 = Vp;   // dead after k_av
  float* Tmp2 = Sc;   // dead after k_av/k_argmax
  float* Hbuf = Qp;   // Val0 dead after Wo GEMM

  k_embed_qkv<<<R / 16, 256, 0, stream>>>(x, sval, svec, tval, tvec, Wq, bq, Wk,
                                          bk, Wv, bv, Qp, Kp, Vp);
  k_qk_m<<<(HBT * Nn) / 8, 256, 0, stream>>>(Qp, Kp, adj, Wp, bp, Mb);
  k_topk<<<HBT, 256, 0, stream>>>(Mb, Mtop);
  k_scores<<<dim3(HBT, 4, 8), 256, 0, stream>>>(Qp, Kp, Mtop, Sc);
  k_softmax<<<HBT * Sp, 256, 0, stream>>>(Sc);
  k_argmax<<<HBT * Nn / 256, 256, 0, stream>>>(Sc, Cp);
  k_av<<<dim3(HBT, 4), 256, 0, stream>>>(Sc, Vp, Av4);
  k_gather<<<R * Ff / 256, 256, 0, stream>>>(Av4, Cp, Val0);
  k_gemm<0><<<R / 16, 256, 0, stream>>>(Val0, Wo, bo, nullptr, Tmp1);
  k_ln<<<R / 8, 256, 0, stream>>>(Tmp1, lng, lnb, Val1, 1);
  k_gemm<1><<<R / 16, 256, 0, stream>>>(Val1, W1, b1, nullptr, Tmp2);
  k_gemm<2><<<R / 16, 256, 0, stream>>>(Tmp2, W2, b2, Val1, Hbuf);
  k_ln<<<R / 8, 256, 0, stream>>>(Hbuf, nullptr, nullptr, out, 0);
}
